// GroupedSwiGLUExperts_89910845375253
// MI455X (gfx1250) — matrix operand order repaired and checked by reading, not compiled, not run on hardware
//
#include <hip/hip_runtime.h>

typedef __attribute__((ext_vector_type(16))) __bf16 v16bf;
typedef __attribute__((ext_vector_type(8)))  __bf16 v8bf;
typedef __attribute__((ext_vector_type(8)))  float  v8f;

#define ME   2048   // tokens
#define KK   2048   // hidden dim
#define NN   1408   // intermediate dim
#define EE   8      // experts
#define TOPK 2
#define CAP  1024   // capacity per expert
#define RR   (ME * TOPK)

#define BM   128    // rows per block tile
#define BN   64     // output cols per block tile
#define KB   32     // WMMA K depth per step
#define LDT  40     // padded LDS row stride (elements); 80B rows keep 16B align

// ---------------------------------------------------------------------------
__global__ void moe_zero_out(float4* __restrict__ out, int n4) {
    int i = blockIdx.x * blockDim.x + threadIdx.x;
    if (i < n4) out[i] = make_float4(0.f, 0.f, 0.f, 0.f);
}

// ---------------------------------------------------------------------------
// Routing: wave w == expert w scans all R rows in original order; ballot-based
// prefix sums reproduce the reference's stable-sort positions. Deterministic.
__global__ __launch_bounds__(256) void moe_route(
    const int*   __restrict__ flat_idx,
    const float* __restrict__ flat_gate,
    int*   __restrict__ rowcnt,
    int*   __restrict__ tok_of,
    float* __restrict__ gate_of)
{
    const int w    = threadIdx.x >> 5;   // expert id (8 waves)
    const int lane = threadIdx.x & 31;
    int cnt = 0;
    for (int base = 0; base < RR; base += 32) {
        const int r = base + lane;
        const bool match = (flat_idx[r] == w);
        const unsigned long long mask = __ballot(match);
        const int prefix = __popcll(mask & ((1ull << lane) - 1ull));
        if (match) {
            const int pos = cnt + prefix;
            if (pos < CAP) {
                tok_of[w * CAP + pos]  = r >> 1;         // TOPK == 2
                gate_of[w * CAP + pos] = flat_gate[r];
            }
        }
        cnt += __popcll(mask);
    }
    if (lane == 0) rowcnt[w] = (cnt < CAP) ? cnt : CAP;
}

// ---------------------------------------------------------------------------
// Fragment helpers (16-bit WMMA VGPR layouts per CDNA5 ISA 7.12.2):
// A 16x32: lane L holds row M=L%16; lanes<16 K-chunks {0..7,16..23},
//          lanes>=16 K-chunks {8..15,24..31}.
// B 32x16: lane L holds col N=L%16; same K-chunks as A.
__device__ inline v16bf frag_A(const __bf16* As, int arow, int half) {
    const v8bf a0 = *reinterpret_cast<const v8bf*>(&As[arow * LDT + half * 8]);
    const v8bf a1 = *reinterpret_cast<const v8bf*>(&As[arow * LDT + 16 + half * 8]);
    v16bf a;
    for (int i = 0; i < 8; ++i) { a[i] = a0[i]; a[8 + i] = a1[i]; }
    return a;
}
__device__ inline v16bf frag_B(const __bf16* Bs, int brow, int half) {
    const v8bf b0 = *reinterpret_cast<const v8bf*>(&Bs[brow * LDT + half * 8]);
    const v8bf b1 = *reinterpret_cast<const v8bf*>(&Bs[brow * LDT + 16 + half * 8]);
    v16bf b;
    for (int i = 0; i < 8; ++i) { b[i] = b0[i]; b[8 + i] = b1[i]; }
    return b;
}

// Pack 8 f32 -> v8bf (lowers to v_cvt_pk_bf16_f32 pairs).
__device__ inline v8bf cvt8(const float4 x, const float4 y) {
    v8bf r;
    r[0] = (__bf16)x.x; r[1] = (__bf16)x.y; r[2] = (__bf16)x.z; r[3] = (__bf16)x.w;
    r[4] = (__bf16)y.x; r[5] = (__bf16)y.y; r[6] = (__bf16)y.z; r[7] = (__bf16)y.w;
    return r;
}

// ---------------------------------------------------------------------------
// GEMM1: g = A @ gateW^T, u = A @ upW^T (fused), SwiGLU -> Hbuf (bf16).
// Register double-buffered staging: next tile's global b128 loads issue
// before the WMMA block so HBM latency overlaps matrix math.
__global__ __launch_bounds__(256) void moe_gemm1(
    const float* __restrict__ flat_h,
    const float* __restrict__ gate_w,
    const float* __restrict__ up_w,
    const int*   __restrict__ rowcnt,
    const int*   __restrict__ tok_of,
    unsigned short* __restrict__ hbuf_raw)
{
    __shared__ __align__(16) __bf16 As[BM * LDT];
    __shared__ __align__(16) __bf16 Bg[BN * LDT];
    __shared__ __align__(16) __bf16 Bu[BN * LDT];

    __bf16* __restrict__ Hbuf = reinterpret_cast<__bf16*>(hbuf_raw);

    const int e   = blockIdx.z;
    const int mb  = blockIdx.y * BM;
    const int n0  = blockIdx.x * BN;
    const int cnt = rowcnt[e];
    if (mb >= cnt) return;

    const int tid  = threadIdx.x;
    const int wv   = tid >> 5;
    const int lane = tid & 31;
    const int col  = lane & 15;
    const int half = lane >> 4;
    const int rw   = (wv & 3) * 32;   // wave's 32-row strip within BM
    const int nw   = (wv >> 2) * 32;  // wave's 32-col strip within BN

    // A: thread owns (row ar, 16-wide K chunk ak); 4x b128 loads + 2x b128 DS.
    const int ar = tid >> 1;
    const int ak = (tid & 1) * 16;
    const int arow_g = mb + ar;
    const int atok = (arow_g < cnt) ? tok_of[e * CAP + arow_g] : -1;
    const float* __restrict__ aSrc =
        flat_h + (size_t)((atok >= 0) ? atok : 0) * KK + ak;

    // B: thread owns (row br, 8-wide K chunk bk); 2x b128 loads + 1x b128 DS.
    const int br = tid >> 2;
    const int bk = (tid & 3) * 8;
    const float* __restrict__ gSrc = gate_w + ((size_t)e * NN + n0 + br) * KK + bk;
    const float* __restrict__ uSrc = up_w   + ((size_t)e * NN + n0 + br) * KK + bk;

    float4 ra[4], rg[2], ru[2];
    auto loadTile = [&](int kb) {
        ra[0] = *(const float4*)(aSrc + kb);
        ra[1] = *(const float4*)(aSrc + kb + 4);
        ra[2] = *(const float4*)(aSrc + kb + 8);
        ra[3] = *(const float4*)(aSrc + kb + 12);
        if (atok < 0) {
            ra[0] = ra[1] = ra[2] = ra[3] = make_float4(0.f, 0.f, 0.f, 0.f);
        }
        rg[0] = *(const float4*)(gSrc + kb);
        rg[1] = *(const float4*)(gSrc + kb + 4);
        ru[0] = *(const float4*)(uSrc + kb);
        ru[1] = *(const float4*)(uSrc + kb + 4);
    };
    auto storeTile = [&]() {
        *reinterpret_cast<v8bf*>(&As[ar * LDT + ak])     = cvt8(ra[0], ra[1]);
        *reinterpret_cast<v8bf*>(&As[ar * LDT + ak + 8]) = cvt8(ra[2], ra[3]);
        *reinterpret_cast<v8bf*>(&Bg[br * LDT + bk])     = cvt8(rg[0], rg[1]);
        *reinterpret_cast<v8bf*>(&Bu[br * LDT + bk])     = cvt8(ru[0], ru[1]);
    };

    v8f accg[2][2] = {};
    v8f accu[2][2] = {};

    loadTile(0);
    for (int kb = 0; kb < KK; kb += KB) {
        storeTile();
        __syncthreads();
        if (kb + KB < KK) {
            loadTile(kb + KB);                       // overlap HBM with WMMA
            __builtin_prefetch(gSrc + kb + 2 * KB, 0, 1);  // global_prefetch_b8
        }
        const v16bf a0 = frag_A(As, rw + col, half);
        const v16bf a1 = frag_A(As, rw + 16 + col, half);
        for (int jn = 0; jn < 2; ++jn) {
            const v16bf bg = frag_B(Bg, nw + jn * 16 + col, half);
            accg[0][jn] = __builtin_amdgcn_wmma_f32_16x16x32_bf16(
                false, a0, false, bg, (short)0, accg[0][jn], false, false);
            accg[1][jn] = __builtin_amdgcn_wmma_f32_16x16x32_bf16(
                false, a1, false, bg, (short)0, accg[1][jn], false, false);
            const v16bf bu = frag_B(Bu, nw + jn * 16 + col, half);
            accu[0][jn] = __builtin_amdgcn_wmma_f32_16x16x32_bf16(
                false, a0, false, bu, (short)0, accu[0][jn], false, false);
            accu[1][jn] = __builtin_amdgcn_wmma_f32_16x16x32_bf16(
                false, a1, false, bu, (short)0, accu[1][jn], false, false);
        }
        __syncthreads();
    }

    // Epilogue: SwiGLU with reference clamps; store h as bf16.
    for (int im = 0; im < 2; ++im) {
        for (int v = 0; v < 8; ++v) {
            const int row  = rw + im * 16 + half * 8 + v;   // C/D: M = v (+8 hi)
            const int grow = mb + row;
            if (grow >= cnt) continue;
            const size_t ob = ((size_t)e * CAP + grow) * NN + n0 + nw;
            for (int jn = 0; jn < 2; ++jn) {
                float g = accg[im][jn][v];
                g = fminf(g, 10.f);
                const float s = g / (1.f + __expf(-g));            // silu
                const float u = fminf(fmaxf(accu[im][jn][v], -10.f), 10.f);
                Hbuf[ob + jn * 16 + col] = (__bf16)(s * u);
            }
        }
    }
}

// ---------------------------------------------------------------------------
// GEMM2: d = H @ downW^T over inner dim N; gated atomic scatter-add into out.
// A-tile (already bf16 in global) is staged with the CDNA5 async DMA path:
// GLOBAL_LOAD_ASYNC_TO_LDS_B128 tracked by ASYNCcnt — no VGPR round-trip.
__global__ __launch_bounds__(256) void moe_gemm2(
    const unsigned short* __restrict__ hbuf_raw,
    const float* __restrict__ down_w,
    const int*   __restrict__ rowcnt,
    const int*   __restrict__ tok_of,
    const float* __restrict__ gate_of,
    float* __restrict__ out)
{
    __shared__ __align__(16) __bf16 As[BM * LDT];
    __shared__ __align__(16) __bf16 Bs[BN * LDT];

    const __bf16* __restrict__ Hbuf = reinterpret_cast<const __bf16*>(hbuf_raw);

    const int e   = blockIdx.z;
    const int mb  = blockIdx.y * BM;
    const int c0  = blockIdx.x * BN;       // output hidden-dim columns
    const int cnt = rowcnt[e];
    if (mb >= cnt) return;

    const int tid  = threadIdx.x;
    const int wv   = tid >> 5;
    const int lane = tid & 31;
    const int col  = lane & 15;
    const int half = lane >> 4;
    const int rw   = (wv & 3) * 32;
    const int nw   = (wv >> 2) * 32;

    // A: thread owns (row ar, 16-wide chunk ak) -> async copy of 32 bytes.
    const int ar = tid >> 1;
    const int ak = (tid & 1) * 16;
    const __bf16* __restrict__ hSrc =
        Hbuf + ((size_t)e * CAP + mb + ar) * NN + ak;
    // LDS byte address = low 32 bits of the generic pointer (aperture in hi).
    const unsigned ldsA = (unsigned)(size_t)&As[ar * LDT + ak];

    // B: thread owns (row br, 8-wide chunk bk) of down weights (f32).
    const int br = tid >> 2;
    const int bk = (tid & 3) * 8;
    const float* __restrict__ dSrc = down_w + ((size_t)e * KK + c0 + br) * NN + bk;

    float4 rd[2];
    auto loadB = [&](int nb) {
        rd[0] = *(const float4*)(dSrc + nb);
        rd[1] = *(const float4*)(dSrc + nb + 4);
    };
    auto storeB = [&]() {
        *reinterpret_cast<v8bf*>(&Bs[br * LDT + bk]) = cvt8(rd[0], rd[1]);
    };
    // Async DMA of this thread's 32-byte A chunk; INST_OFFSET applies to both
    // LDS and global addresses (ISA 10.x async addressing), so one address
    // pair covers both 16B halves.
    auto asyncA = [&](int nb) {
        const unsigned long long ga = (unsigned long long)(size_t)(hSrc + nb);
        asm volatile("global_load_async_to_lds_b128 %0, %1, off"
                     :: "v"(ldsA), "v"(ga) : "memory");
        asm volatile("global_load_async_to_lds_b128 %0, %1, off offset:16"
                     :: "v"(ldsA), "v"(ga) : "memory");
    };

    v8f acc[2][2] = {};

    loadB(0);
    asyncA(0);
    for (int nb = 0; nb < NN; nb += KB) {
        storeB();
        asm volatile("s_wait_asynccnt 0x0" ::: "memory");   // A tile landed
        __syncthreads();
        if (nb + KB < NN) {
            loadB(nb + KB);
            __builtin_prefetch(dSrc + nb + 2 * KB, 0, 1);
        }
        const v16bf a0 = frag_A(As, rw + col, half);
        const v16bf a1 = frag_A(As, rw + 16 + col, half);
        for (int jn = 0; jn < 2; ++jn) {
            const v16bf b = frag_B(Bs, nw + jn * 16 + col, half);
            acc[0][jn] = __builtin_amdgcn_wmma_f32_16x16x32_bf16(
                false, a0, false, b, (short)0, acc[0][jn], false, false);
            acc[1][jn] = __builtin_amdgcn_wmma_f32_16x16x32_bf16(
                false, a1, false, b, (short)0, acc[1][jn], false, false);
        }
        __syncthreads();                 // readers done -> safe to DMA next tile
        if (nb + KB < NN) asyncA(nb + KB);
    }

    // Gated combine: out[tok] += gate * d_row (global_atomic_add_f32).
    for (int im = 0; im < 2; ++im) {
        for (int v = 0; v < 8; ++v) {
            const int row  = rw + im * 16 + half * 8 + v;
            const int grow = mb + row;
            if (grow >= cnt) continue;
            const int   tok  = tok_of[e * CAP + grow];
            const float gate = gate_of[e * CAP + grow];
            for (int jn = 0; jn < 2; ++jn) {
                atomicAdd(&out[(size_t)tok * KK + c0 + nw + jn * 16 + col],
                          gate * acc[im][jn][v]);
            }
        }
    }
}

// ---------------------------------------------------------------------------
extern "C" void kernel_launch(void* const* d_in, const int* in_sizes, int n_in,
                              void* d_out, int out_size, void* d_ws, size_t ws_size,
                              hipStream_t stream) {
    (void)in_sizes; (void)n_in; (void)out_size; (void)ws_size;

    const float* flat_h    = (const float*)d_in[0];   // (M,K) f32
    const int*   flat_idx  = (const int*)  d_in[1];   // (M,TOPK) i32
    const float* flat_gate = (const float*)d_in[2];   // (M,TOPK) f32
    const float* gate_w    = (const float*)d_in[3];   // (E,N,K) f32
    const float* up_w      = (const float*)d_in[4];   // (E,N,K) f32
    const float* down_w    = (const float*)d_in[5];   // (E,K,N) f32
    float* out = (float*)d_out;                        // (M,K) f32

    // Workspace layout
    char* ws = (char*)d_ws;
    int*   rowcnt  = (int*)ws;                                     // E ints
    int*   tok_of  = (int*)(ws + 256);                             // E*CAP ints
    float* gate_of = (float*)(ws + 256 + (size_t)EE * CAP * 4);    // E*CAP f32
    unsigned short* hbuf =
        (unsigned short*)(ws + 256 + (size_t)2 * EE * CAP * 4);    // E*CAP*N bf16

    const int n4 = (ME * KK) / 4;
    moe_zero_out<<<(n4 + 255) / 256, 256, 0, stream>>>((float4*)out, n4);

    moe_route<<<1, 256, 0, stream>>>(flat_idx, flat_gate, rowcnt, tok_of, gate_of);

    dim3 g1(NN / BN, CAP / BM, EE);   // (22, 8, 8)
    moe_gemm1<<<g1, 256, 0, stream>>>(flat_h, gate_w, up_w, rowcnt, tok_of, hbuf);

    dim3 g2(KK / BN, CAP / BM, EE);   // (32, 8, 8)
    moe_gemm2<<<g2, 256, 0, stream>>>(hbuf, down_w, rowcnt, tok_of, gate_of, out);
}
